// CompressedCausalAttention_60249801228500
// MI455X (gfx1250) — matrix operand order repaired and checked by reading, not compiled, not run on hardware
//
#include <hip/hip_runtime.h>
#include <hip/hip_bf16.h>

// Problem constants (match reference)
#define S_LEN  2048
#define BATCH  2
#define CMODEL 512
#define NHEAD  8
#define HDIM   64
#define D3     (3 * CMODEL)

#define XELEMS  (S_LEN * BATCH * CMODEL)   // 2,097,152
#define WQELEMS (D3 * CMODEL)              //   786,432
#define WCELEMS (CMODEL * CMODEL)          //   262,144

typedef __attribute__((ext_vector_type(16))) _Float16 v16h;
typedef __attribute__((ext_vector_type(8)))  _Float16 v8h;
typedef __attribute__((ext_vector_type(4)))  _Float16 v4h;
typedef __attribute__((ext_vector_type(8)))  float    v8f;
typedef __attribute__((ext_vector_type(4)))  float    v4f;

union HFrag {
    v16h     v;
    v8h      h[2];
    _Float16 e[16];
};

static __device__ __forceinline__ v8f wmma_f16(const v16h& a, const v16h& b, const v8f& c) {
    // D = A(16x32 f16) * B(32x16 f16) + C(16x16 f32)
    return __builtin_amdgcn_wmma_f32_16x16x32_f16(false, a, false, b, (short)0, c, false, false);
}

// ---------------------------------------------------------------------------
// Kernel 0: one-shot f32 -> f16 materialization.
//   xh    = f16(x + pe)      (S*B, C) row-major
//   wqkvh = f16(Wqkv)        (3C, C)  row-major
//   wch   = f16(Wc)          (C, C)   row-major
// ---------------------------------------------------------------------------
__global__ __launch_bounds__(256) void prep_kernel(
        const float* __restrict__ x, const float* __restrict__ pe,
        const float* __restrict__ Wqkv, const float* __restrict__ Wc,
        _Float16* __restrict__ xh, _Float16* __restrict__ wqkvh,
        _Float16* __restrict__ wch) {
    const int i4 = blockIdx.x * 256 + threadIdx.x;   // one float4 per thread
    const int X4 = XELEMS / 4, Q4 = WQELEMS / 4;
    if (i4 < X4) {
        const v4f a = *(const v4f*)(x + (size_t)i4 * 4);
        const v4f p = *(const v4f*)(pe + (size_t)i4 * 4);
        v4h o;
#pragma unroll
        for (int i = 0; i < 4; ++i) o[i] = (_Float16)(a[i] + p[i]);
        *(v4h*)(xh + (size_t)i4 * 4) = o;
    } else if (i4 < X4 + Q4) {
        const int j = i4 - X4;
        const v4f w = *(const v4f*)(Wqkv + (size_t)j * 4);
        v4h o;
#pragma unroll
        for (int i = 0; i < 4; ++i) o[i] = (_Float16)w[i];
        *(v4h*)(wqkvh + (size_t)j * 4) = o;
    } else {
        const int j = i4 - X4 - Q4;
        const v4f w = *(const v4f*)(Wc + (size_t)j * 4);
        v4h o;
#pragma unroll
        for (int i = 0; i < 4; ++i) o[i] = (_Float16)w[i];
        *(v4h*)(wch + (size_t)j * 4) = o;
    }
}

// ---------------------------------------------------------------------------
// Kernel 1: QKV projection.  M = S*B = 4096, N = 3C = 1536, K = C = 512.
// One wave computes a 16(M) x 64(N) strip with a double-buffered K loop:
// fragments for chunk k+32 are in flight while chunk k is multiplied.
// Outputs f16: q,k as (b,h,s,64); v transposed as (b,h,64,s).
// ---------------------------------------------------------------------------
__global__ __launch_bounds__(128) void qkv_gemm_kernel(
        const _Float16* __restrict__ xh, const _Float16* __restrict__ wqkvh,
        const float* __restrict__ bqkv,
        _Float16* __restrict__ qbuf, _Float16* __restrict__ kbuf,
        _Float16* __restrict__ vT) {
    const int lane  = threadIdx.x & 31;
    const int wave  = threadIdx.x >> 5;
    const int tile  = blockIdx.x * 4 + wave;
    const int Ngrp  = D3 / 64;                  // 24 groups of 4 N-tiles
    const int mBase = (tile / Ngrp) * 16;
    const int nBase = (tile % Ngrp) * 64;
    const int grp   = lane >> 4, ln = lane & 15;

    const _Float16* arp = xh + (size_t)(mBase + ln) * CMODEL;
    const _Float16* wp0 = wqkvh + (size_t)(nBase + ln) * CMODEL + grp * 16;

    auto loadA = [&](HFrag& f, int kc) {
        f.h[0] = *(const v8h*)(arp + kc + grp * 16);
        f.h[1] = *(const v8h*)(arp + kc + 8 + grp * 16);
    };
    auto loadB = [&](v16h* bf, int kc) {
#pragma unroll
        for (int j = 0; j < 4; ++j)
            bf[j] = *(const v16h*)(wp0 + (size_t)j * 16 * CMODEL + kc);
    };

    v8f acc[4] = {{}, {}, {}, {}};
    HFrag aA, aB;
    v16h  bA[4], bB[4];
    loadA(aA, 0);
    loadB(bA, 0);
    for (int kc = 0; kc < CMODEL; kc += 64) {
        loadA(aB, kc + 32);                       // next chunk in flight
        loadB(bB, kc + 32);
        __builtin_prefetch(arp + kc + 64, 0, 3);  // near-cache streaming hint
#pragma unroll
        for (int j = 0; j < 4; ++j) acc[j] = wmma_f16(aA.v, bA[j], acc[j]);
        if (kc + 64 < CMODEL) {
            loadA(aA, kc + 64);
            loadB(bA, kc + 64);
        }
#pragma unroll
        for (int j = 0; j < 4; ++j) acc[j] = wmma_f16(aB.v, bB[j], acc[j]);
    }

    // epilogue: bias + scatter.  nBase is 64-aligned -> whole strip is inside
    // one of the q/k/v 512-wide segments, so the branch is wave-uniform.
#pragma unroll
    for (int j = 0; j < 4; ++j) {
        const int dcol = nBase + j * 16 + ln;
        const float bias = bqkv[dcol];
#pragma unroll
        for (int v = 0; v < 8; ++v) {
            const int row = mBase + grp * 8 + v;   // s*B + b
            const int s = row >> 1;                // BATCH == 2
            const int b = row & 1;
            const _Float16 hv = (_Float16)(acc[j][v] + bias);
            if (dcol < CMODEL) {
                const int h = dcol >> 6, cc = dcol & 63;
                qbuf[(((size_t)(b * NHEAD + h) * S_LEN) + s) * HDIM + cc] = hv;
            } else if (dcol < 2 * CMODEL) {
                const int d2 = dcol - CMODEL, h = d2 >> 6, cc = d2 & 63;
                kbuf[(((size_t)(b * NHEAD + h) * S_LEN) + s) * HDIM + cc] = hv;
            } else {
                const int d2 = dcol - 2 * CMODEL, h = d2 >> 6, cc = d2 & 63;
                vT[(((size_t)(b * NHEAD + h) * HDIM) + cc) * S_LEN + s] = hv;
            }
        }
    }
}

// ---------------------------------------------------------------------------
// Kernel 2: flash attention.  One wave per (b,h) and 32 query rows:
// two 16-row S-tiles share every K and V fragment.  V fragments are loaded
// *before* the softmax VALU section so their latency hides under it.
// ---------------------------------------------------------------------------
__global__ __launch_bounds__(32) void flash_attn_kernel(
        const _Float16* __restrict__ qbuf, const _Float16* __restrict__ kbuf,
        const _Float16* __restrict__ vT, _Float16* __restrict__ attnOut) {
    __shared__ alignas(16) _Float16 P[2][16 * 40];  // two 16x32 P tiles, stride 40

    const int lane  = threadIdx.x & 31;
    const int grp   = lane >> 4, ln = lane & 15;
    const int sTile = blockIdx.x & (S_LEN / 32 - 1);   // 64 tiles of 32 rows
    const int bh    = blockIdx.x >> 6;                 // b*NHEAD + h
    const int sBase = sTile * 32;

    const _Float16* qh = qbuf + (size_t)bh * S_LEN * HDIM;
    const _Float16* kh = kbuf + (size_t)bh * S_LEN * HDIM;
    const _Float16* vh = vT   + (size_t)bh * HDIM * S_LEN;

    // Q A-fragments for both row tiles, K chunks c in [0,32) and [32,64)
    HFrag aq[2][2];
#pragma unroll
    for (int r = 0; r < 2; ++r) {
        const _Float16* qr = qh + (size_t)(sBase + r * 16 + ln) * HDIM;
        aq[r][0].h[0] = *(const v8h*)(qr + grp * 16);
        aq[r][0].h[1] = *(const v8h*)(qr + 8 + grp * 16);
        aq[r][1].h[0] = *(const v8h*)(qr + 32 + grp * 16);
        aq[r][1].h[1] = *(const v8h*)(qr + 40 + grp * 16);
    }

    float m[2][8], l[2][8];
    v8f O[2][4];
    const v8f vzero = {};
#pragma unroll
    for (int r = 0; r < 2; ++r)
#pragma unroll
        for (int v = 0; v < 8; ++v) { m[r][v] = -__builtin_inff(); l[r][v] = 0.0f; }
#pragma unroll
    for (int r = 0; r < 2; ++r)
#pragma unroll
        for (int j = 0; j < 4; ++j) O[r][j] = vzero;

    const float kLog2e = 1.44269504088896f;
    const int nkb = sTile + 1;                  // 32-wide key blocks incl. diagonal
    for (int kb = 0; kb < nkb; ++kb) {
        const int tB = kb * 32;
        // shared K B-fragments (lane = column t, 16 consecutive c per lane-half)
        const _Float16* kr0 = kh + (size_t)(tB + ln) * HDIM;
        const _Float16* kr1 = kr0 + 16 * HDIM;
        const v16h bk0l = *(const v16h*)(kr0 + grp * 16);
        const v16h bk0h = *(const v16h*)(kr0 + 32 + grp * 16);
        const v16h bk1l = *(const v16h*)(kr1 + grp * 16);
        const v16h bk1h = *(const v16h*)(kr1 + 32 + grp * 16);

        // V B-fragments hoisted: latency hides under QK^T + softmax VALU work
        const int tOff = tB + grp * 16;
        v16h bv[4];
#pragma unroll
        for (int j = 0; j < 4; ++j)
            bv[j] = *(const v16h*)(vh + (size_t)(j * 16 + ln) * S_LEN + tOff);

        if (kb + 1 < nkb) {                     // near-cache hint for next block
            __builtin_prefetch(kr0 + 32 * HDIM, 0, 3);
            __builtin_prefetch(kr1 + 32 * HDIM, 0, 3);
            __builtin_prefetch(vh + (size_t)ln * S_LEN + tOff + 32, 0, 3);
            __builtin_prefetch(vh + (size_t)(48 + ln) * S_LEN + tOff + 32, 0, 3);
        }

        const int col0 = tB + ln, col1 = col0 + 16;
#pragma unroll
        for (int r = 0; r < 2; ++r) {
            v8f s0 = vzero, s1 = vzero;
            s0 = wmma_f16(aq[r][0].v, bk0l, s0);
            s0 = wmma_f16(aq[r][1].v, bk0h, s0);
            s1 = wmma_f16(aq[r][0].v, bk1l, s1);
            s1 = wmma_f16(aq[r][1].v, bk1h, s1);

            const int rowBase = sBase + r * 16;
            const bool needMask = (tB + 31 > rowBase);   // wave-uniform
            float rmax[8], rsum[8];
#pragma unroll
            for (int v = 0; v < 8; ++v) {
                const int row = rowBase + grp * 8 + v;
                float a0 = s0[v] * 0.125f;               // 1/sqrt(64)
                float a1 = s1[v] * 0.125f;
                if (needMask) {
                    if (col0 > row) a0 = -__builtin_inff();
                    if (col1 > row) a1 = -__builtin_inff();
                }
                s0[v] = a0; s1[v] = a1;
                rmax[v] = fmaxf(a0, a1);
            }
#pragma unroll
            for (int off = 8; off >= 1; off >>= 1)
#pragma unroll
                for (int v = 0; v < 8; ++v)
                    rmax[v] = fmaxf(rmax[v], __shfl_xor(rmax[v], off, 32));

            float corr[8];
#pragma unroll
            for (int v = 0; v < 8; ++v) {
                const float newm = fmaxf(m[r][v], rmax[v]);
                corr[v] = exp2f((m[r][v] - newm) * kLog2e);
                m[r][v] = newm;
                const float p0 = exp2f((s0[v] - newm) * kLog2e);
                const float p1 = exp2f((s1[v] - newm) * kLog2e);
                s0[v] = p0; s1[v] = p1;
                rsum[v] = p0 + p1;
            }
#pragma unroll
            for (int off = 8; off >= 1; off >>= 1)
#pragma unroll
                for (int v = 0; v < 8; ++v)
                    rsum[v] += __shfl_xor(rsum[v], off, 32);
#pragma unroll
            for (int v = 0; v < 8; ++v) {
                l[r][v] = l[r][v] * corr[v] + rsum[v];
#pragma unroll
                for (int j = 0; j < 4; ++j) O[r][j][v] *= corr[v];
            }
            // P tile r through LDS: C/D layout -> A layout
#pragma unroll
            for (int v = 0; v < 8; ++v) {
                const int rr = grp * 8 + v;
                P[r][rr * 40 + ln]      = (_Float16)s0[v];
                P[r][rr * 40 + 16 + ln] = (_Float16)s1[v];
            }
        }
        __syncthreads();
        HFrag pa[2];
#pragma unroll
        for (int r = 0; r < 2; ++r) {
            const _Float16* pr = &P[r][ln * 40];
            pa[r].h[0] = *(const v8h*)(pr + grp * 16);
            pa[r].h[1] = *(const v8h*)(pr + 8 + grp * 16);
        }
        // PV: each (already resident) V fragment feeds both row tiles
#pragma unroll
        for (int j = 0; j < 4; ++j) {
            O[0][j] = wmma_f16(pa[0].v, bv[j], O[0][j]);
            O[1][j] = wmma_f16(pa[1].v, bv[j], O[1][j]);
        }
        __syncthreads();
    }

    // normalize and store f16 attention output, layout (s, b, h*64+c)
    const int b = bh >> 3, h = bh & 7;
#pragma unroll
    for (int r = 0; r < 2; ++r)
#pragma unroll
        for (int v = 0; v < 8; ++v) {
            const float inv = 1.0f / l[r][v];
            const int s = sBase + r * 16 + grp * 8 + v;
            _Float16* orow = attnOut + ((size_t)s * BATCH + b) * CMODEL + h * HDIM;
#pragma unroll
            for (int j = 0; j < 4; ++j)
                orow[j * 16 + ln] = (_Float16)(O[r][j][v] * inv);
        }
}

// ---------------------------------------------------------------------------
// Kernel 3: output projection.  M = 4096, N = 512, K = 512.
// 16(M) x 64(N) strip per wave, double-buffered like kernel 1.
// ---------------------------------------------------------------------------
__global__ __launch_bounds__(128) void out_gemm_kernel(
        const _Float16* __restrict__ attnOut, const _Float16* __restrict__ wch,
        const float* __restrict__ bc, float* __restrict__ out) {
    const int lane  = threadIdx.x & 31;
    const int wave  = threadIdx.x >> 5;
    const int tile  = blockIdx.x * 4 + wave;
    const int Ngrp  = CMODEL / 64;              // 8
    const int mBase = (tile / Ngrp) * 16;
    const int nBase = (tile % Ngrp) * 64;
    const int grp   = lane >> 4, ln = lane & 15;

    const _Float16* arp = attnOut + (size_t)(mBase + ln) * CMODEL;
    const _Float16* wp0 = wch + (size_t)(nBase + ln) * CMODEL + grp * 16;

    auto loadA = [&](HFrag& f, int kc) {
        f.h[0] = *(const v8h*)(arp + kc + grp * 16);
        f.h[1] = *(const v8h*)(arp + kc + 8 + grp * 16);
    };
    auto loadB = [&](v16h* bf, int kc) {
#pragma unroll
        for (int j = 0; j < 4; ++j)
            bf[j] = *(const v16h*)(wp0 + (size_t)j * 16 * CMODEL + kc);
    };

    v8f acc[4] = {{}, {}, {}, {}};
    HFrag aA, aB;
    v16h  bA[4], bB[4];
    loadA(aA, 0);
    loadB(bA, 0);
    for (int kc = 0; kc < CMODEL; kc += 64) {
        loadA(aB, kc + 32);
        loadB(bB, kc + 32);
        __builtin_prefetch(arp + kc + 64, 0, 3);
#pragma unroll
        for (int j = 0; j < 4; ++j) acc[j] = wmma_f16(aA.v, bA[j], acc[j]);
        if (kc + 64 < CMODEL) {
            loadA(aA, kc + 64);
            loadB(bA, kc + 64);
        }
#pragma unroll
        for (int j = 0; j < 4; ++j) acc[j] = wmma_f16(aB.v, bB[j], acc[j]);
    }
#pragma unroll
    for (int j = 0; j < 4; ++j) {
        const int dcol = nBase + j * 16 + ln;
        const float bias = bc[dcol];
#pragma unroll
        for (int v = 0; v < 8; ++v) {
            const int row = mBase + grp * 8 + v;
            out[(size_t)row * CMODEL + dcol] = acc[j][v] + bias;
        }
    }
}

// ---------------------------------------------------------------------------
extern "C" void kernel_launch(void* const* d_in, const int* in_sizes, int n_in,
                              void* d_out, int out_size, void* d_ws, size_t ws_size,
                              hipStream_t stream) {
    (void)in_sizes; (void)n_in; (void)out_size; (void)ws_size;
    const float* x    = (const float*)d_in[0];
    const float* pe   = (const float*)d_in[1];
    // d_in[2] = content_mask (recomputed analytically), d_in[3] = pad (unused)
    const float* Wqkv = (const float*)d_in[4];
    const float* bqkv = (const float*)d_in[5];
    const float* Wc   = (const float*)d_in[6];
    const float* bc   = (const float*)d_in[7];
    float* out = (float*)d_out;

    char* ws = (char*)d_ws;
    const size_t headBytes = (size_t)BATCH * NHEAD * S_LEN * HDIM * sizeof(_Float16); // 4 MiB
    _Float16* qbuf    = (_Float16*)(ws + 0 * headBytes);
    _Float16* kbuf    = (_Float16*)(ws + 1 * headBytes);
    _Float16* vT      = (_Float16*)(ws + 2 * headBytes);
    _Float16* attnOut = (_Float16*)(ws + 3 * headBytes);
    _Float16* xh      = (_Float16*)(ws + 4 * headBytes);                 // 4 MiB
    _Float16* wqkvh   = (_Float16*)(ws + 5 * headBytes);                 // 1.5 MiB
    _Float16* wch     = (_Float16*)(ws + 5 * headBytes + (size_t)WQELEMS * 2);

    // K0: f32 -> f16 materialization (x+pe fused), exact-size grid
    prep_kernel<<<dim3((XELEMS + WQELEMS + WCELEMS) / 4 / 256), 256, 0, stream>>>(
        x, pe, Wqkv, Wc, xh, wqkvh, wch);

    // K1: 256 M-tiles x 24 N-groups, 4 waves (strips) per 128-thread block
    qkv_gemm_kernel<<<dim3((S_LEN * BATCH / 16) * (D3 / 64) / 4), 128, 0, stream>>>(
        xh, wqkvh, bqkv, qbuf, kbuf, vT);

    // K2: one wave per (b, h, 32-query-row tile)
    flash_attn_kernel<<<dim3(BATCH * NHEAD * (S_LEN / 32)), 32, 0, stream>>>(
        qbuf, kbuf, vT, attnOut);

    // K3: 256 M-tiles x 8 N-groups, 4 waves per block
    out_gemm_kernel<<<dim3((S_LEN * BATCH / 16) * (CMODEL / 64) / 4), 128, 0, stream>>>(
        attnOut, wch, bc, out);
}
